// LigerLMHeadGRPO_13554916786397
// MI455X (gfx1250) — matrix operand order repaired and checked by reading, not compiled, not run on hardware
//
#include <hip/hip_runtime.h>
#include <hip/hip_bf16.h>

// ---------------- problem constants ----------------
#define Bb    4
#define Tt    512
#define BT    2048          // B*T tokens
#define Hh    2048
#define Vv    32000
#define BETA  0.1f
#define TEMP  1.0f

// ---------------- GEMM tiling ----------------
#define BLOCK_M 128
#define BLOCK_N 128
#define BLOCK_K 64
#define LDK     (BLOCK_K + 8)      // padded LDS stride (halfs), 144B rows
#define NBLKN   (Vv / BLOCK_N)     // 250
#define NSPLIT  (NBLKN * 2)        // 500: each 128-col block gives 2 64-col splits
#define NSTAGE  (Hh / BLOCK_K)     // 32

typedef __attribute__((ext_vector_type(16))) __bf16 bf16x16;
typedef __attribute__((ext_vector_type(8)))  __bf16 bf16x8;
typedef __attribute__((ext_vector_type(8)))  float  f32x8;

union Frag16 { bf16x16 v; bf16x8 h[2]; };

// ---- CDNA5 async global->LDS path (guarded; sync fallback keeps TU portable) ----
#if __has_builtin(__builtin_amdgcn_global_load_async_to_lds_b128)
#define HAS_ASYNC_LDS 1
#else
#define HAS_ASYNC_LDS 0
#endif

#if HAS_ASYNC_LDS
// builtin expects pointer-to-v4i (GCC vector_size(16) int) in AS1 / AS3
typedef int i32x4_vs __attribute__((vector_size(16)));
typedef __attribute__((address_space(1))) i32x4_vs as1_i32x4;
typedef __attribute__((address_space(3))) i32x4_vs as3_i32x4;
#endif

__device__ __forceinline__ void copy_b128_to_lds(__bf16* lds_dst, const __bf16* gsrc) {
#if HAS_ASYNC_LDS
    // global_load_async_to_lds_b128: ASYNCcnt-tracked, no VGPR staging
    __builtin_amdgcn_global_load_async_to_lds_b128(
        (as1_i32x4*)gsrc, (as3_i32x4*)lds_dst, 0, 0);
#else
    *(bf16x8*)lds_dst = *(const bf16x8*)gsrc;
#endif
}

__device__ __forceinline__ void wait_stage_copies() {
#if HAS_ASYNC_LDS
#if __has_builtin(__builtin_amdgcn_s_wait_asynccnt)
    __builtin_amdgcn_s_wait_asynccnt(0);
#else
    asm volatile("s_wait_asynccnt 0x0" ::: "memory");
#endif
#endif
}

// ---------------- fp32 -> bf16 conversion (streaming) ----------------
__global__ void cvt_f32_to_bf16_kernel(const float* __restrict__ src,
                                       __bf16* __restrict__ dst, int n) {
    int i = (blockIdx.x * blockDim.x + threadIdx.x) * 8;
    int stride = gridDim.x * blockDim.x * 8;
    for (; i < n; i += stride) {
        float4 a = *(const float4*)(src + i);
        float4 b = *(const float4*)(src + i + 4);
        bf16x8 o;
        o[0] = (__bf16)a.x; o[1] = (__bf16)a.y;
        o[2] = (__bf16)a.z; o[3] = (__bf16)a.w;
        o[4] = (__bf16)b.x; o[5] = (__bf16)b.y;
        o[6] = (__bf16)b.z; o[7] = (__bf16)b.w;
        *(bf16x8*)(dst + i) = o;
    }
}

// ------- fused LM-head GEMM + partial (max, sum-exp) per 64-col split -------
// A: (BT, H) bf16 row-major; W: (V, H) bf16 row-major (weight row = B column)
// pmax/psum: (BT, NSPLIT) f32
__global__ __launch_bounds__(256)
void lmhead_partial_kernel(const __bf16* __restrict__ A,
                           const __bf16* __restrict__ W,
                           float* __restrict__ pmax,
                           float* __restrict__ psum) {
    // double-buffered tiles: 2 x (A 18KB + B 18KB) = 72KB of the 320KB WGP LDS
    __shared__ __align__(16) __bf16 As[2][BLOCK_M * LDK];
    __shared__ __align__(16) __bf16 Bs[2][BLOCK_N * LDK];

    const int nblk = blockIdx.x;
    const int mblk = blockIdx.y;
    const int tid  = threadIdx.x;
    const int lane = tid & 31;
    const int wid  = tid >> 5;
    const int wm   = wid & 3;    // 4 wave-rows of 32
    const int wn   = wid >> 2;   // 2 wave-cols of 64

    const int rowBase = mblk * BLOCK_M;
    const int colBase = nblk * BLOCK_N;

    // each of 256 threads copies half a row (32 halfs = 64B) of each tile
    const int lrow  = tid >> 1;
    const int lpart = (tid & 1) * 32;

    const __bf16* gA = A + (size_t)(rowBase + lrow) * Hh + lpart;
    const __bf16* gW = W + (size_t)(colBase + lrow) * Hh + lpart;

    f32x8 acc[2][4];
#pragma unroll
    for (int tm = 0; tm < 2; ++tm)
#pragma unroll
        for (int tn = 0; tn < 4; ++tn)
            acc[tm][tn] = (f32x8)0.0f;

    const int hi  = lane >> 4;   // lane half (WMMA K split)
    const int l15 = lane & 15;

    // current/next tile bases (swapped each stage; keeps the stage loop a
    // single body so accumulators stay in fixed WMMA registers)
    __bf16* curA = As[0]; __bf16* nxtA = As[1];
    __bf16* curB = Bs[0]; __bf16* nxtB = Bs[1];

    auto load_stage_to = [&](int s, __bf16* dA, __bf16* dB) {
        const __bf16* ga = gA + s * BLOCK_K;
        const __bf16* gb = gW + s * BLOCK_K;
        __bf16* sa = dA + lrow * LDK + lpart;
        __bf16* sb = dB + lrow * LDK + lpart;
#pragma unroll
        for (int j = 0; j < 4; ++j)
            copy_b128_to_lds(sa + j * 8, ga + j * 8);
#pragma unroll
        for (int j = 0; j < 4; ++j)
            copy_b128_to_lds(sb + j * 8, gb + j * 8);
    };

    // prologue: fill current buffer
    load_stage_to(0, curA, curB);
    wait_stage_copies();
    __syncthreads();

#pragma unroll 1
    for (int s = 0; s < NSTAGE; ++s) {
        // issue next stage's copies into the other buffer (overlaps with WMMA)
        if (s + 1 < NSTAGE)
            load_stage_to(s + 1, nxtA, nxtB);

#pragma unroll
        for (int kk = 0; kk < BLOCK_K; kk += 32) {
            Frag16 af[2], bf[4];
            // A frag (16x32, M x K): lane<16 -> row l15, K 0-15;
            // lane>=16 -> same row, K 16-31.
#pragma unroll
            for (int tm = 0; tm < 2; ++tm) {
                const __bf16* p = curA + (wm * 32 + tm * 16 + l15) * LDK + kk + hi * 16;
                af[tm].h[0] = *(const bf16x8*)p;
                af[tm].h[1] = *(const bf16x8*)(p + 8);
            }
            // B frag (32x16, K x N): lane<16 -> col l15, K 0-15; lane>=16 -> K 16-31.
#pragma unroll
            for (int tn = 0; tn < 4; ++tn) {
                const __bf16* p = curB + (wn * 64 + tn * 16 + l15) * LDK + kk + hi * 16;
                bf[tn].h[0] = *(const bf16x8*)p;
                bf[tn].h[1] = *(const bf16x8*)(p + 8);
            }
#pragma unroll
            for (int tm = 0; tm < 2; ++tm)
#pragma unroll
                for (int tn = 0; tn < 4; ++tn)
                    acc[tm][tn] = __builtin_amdgcn_wmma_f32_16x16x32_bf16(
                        false, af[tm].v, false, bf[tn].v,
                        (short)0, acc[tm][tn], false, false);
        }
        wait_stage_copies();
        __syncthreads();

        __bf16* t;
        t = curA; curA = nxtA; nxtA = t;
        t = curB; curB = nxtB; nxtB = t;
    }

    // epilogue: per-row max and sum(exp) over this wave's 64 columns
    const float invT = 1.0f / TEMP;
    const int split = nblk * 2 + wn;
#pragma unroll
    for (int tm = 0; tm < 2; ++tm) {
#pragma unroll
        for (int r = 0; r < 8; ++r) {
            float v0 = acc[tm][0][r] * invT;
            float v1 = acc[tm][1][r] * invT;
            float v2 = acc[tm][2][r] * invT;
            float v3 = acc[tm][3][r] * invT;
            float m = fmaxf(fmaxf(v0, v1), fmaxf(v2, v3));
#pragma unroll
            for (int msk = 1; msk <= 8; msk <<= 1)
                m = fmaxf(m, __shfl_xor(m, msk, 32));
            float s = __expf(v0 - m) + __expf(v1 - m) +
                      __expf(v2 - m) + __expf(v3 - m);
#pragma unroll
            for (int msk = 1; msk <= 8; msk <<= 1)
                s += __shfl_xor(s, msk, 32);
            if ((lane & 15) == 0) {
                int row = rowBase + wm * 32 + tm * 16 + (lane >> 4) * 8 + r;
                size_t off = (size_t)row * NSPLIT + split;
                pmax[off] = m;
                psum[off] = s;
            }
        }
    }
}

// -------- selected-token logit: one wave per token, both models --------
__global__ __launch_bounds__(256)
void sel_logit_kernel(const __bf16* __restrict__ Xb, const __bf16* __restrict__ Wb,
                      const __bf16* __restrict__ Rb, const __bf16* __restrict__ WRb,
                      const int* __restrict__ ids, float* __restrict__ selOut) {
    const int lane = threadIdx.x & 31;
    const int wid  = threadIdx.x >> 5;
    const int t = blockIdx.x * (blockDim.x >> 5) + wid;
    if (t >= BT) return;
    const int v = ids[t];
    const __bf16* x0 = Xb + (size_t)t * Hh;
    const __bf16* w0 = Wb + (size_t)v * Hh;
    const __bf16* x1 = Rb + (size_t)t * Hh;
    const __bf16* w1 = WRb + (size_t)v * Hh;
    float a0 = 0.0f, a1 = 0.0f;
    for (int c = lane * 8; c < Hh; c += 32 * 8) {
        bf16x8 xa = *(const bf16x8*)(x0 + c);
        bf16x8 wa = *(const bf16x8*)(w0 + c);
        bf16x8 xb = *(const bf16x8*)(x1 + c);
        bf16x8 wb = *(const bf16x8*)(w1 + c);
#pragma unroll
        for (int j = 0; j < 8; ++j) {
            a0 += (float)xa[j] * (float)wa[j];
            a1 += (float)xb[j] * (float)wb[j];
        }
    }
#pragma unroll
    for (int msk = 1; msk <= 16; msk <<= 1) {
        a0 += __shfl_xor(a0, msk, 32);
        a1 += __shfl_xor(a1, msk, 32);
    }
    if (lane == 0) {
        selOut[t]      = a0 * (1.0f / TEMP);
        selOut[BT + t] = a1 * (1.0f / TEMP);
    }
}

// -------- per-token: merge partials -> logp, ref_logp -> GRPO loss --------
__global__ __launch_bounds__(256)
void combine_kernel(const float* __restrict__ pm0, const float* __restrict__ ps0,
                    const float* __restrict__ pm1, const float* __restrict__ ps1,
                    const float* __restrict__ sel, const float* __restrict__ adv,
                    const int* __restrict__ amask, float* __restrict__ tokenLoss) {
    const int lane = threadIdx.x & 31;
    const int wid  = threadIdx.x >> 5;
    const int t = blockIdx.x * (blockDim.x >> 5) + wid;
    if (t >= BT) return;

    float m0 = -3.4e38f, m1 = -3.4e38f;
    const size_t base = (size_t)t * NSPLIT;
    for (int i = lane; i < NSPLIT; i += 32) {
        m0 = fmaxf(m0, pm0[base + i]);
        m1 = fmaxf(m1, pm1[base + i]);
    }
#pragma unroll
    for (int msk = 1; msk <= 16; msk <<= 1) {
        m0 = fmaxf(m0, __shfl_xor(m0, msk, 32));
        m1 = fmaxf(m1, __shfl_xor(m1, msk, 32));
    }
    float s0 = 0.0f, s1 = 0.0f;
    for (int i = lane; i < NSPLIT; i += 32) {
        s0 += ps0[base + i] * __expf(pm0[base + i] - m0);
        s1 += ps1[base + i] * __expf(pm1[base + i] - m1);
    }
#pragma unroll
    for (int msk = 1; msk <= 16; msk <<= 1) {
        s0 += __shfl_xor(s0, msk, 32);
        s1 += __shfl_xor(s1, msk, 32);
    }
    if (lane == 0) {
        float logp  = sel[t]      - (m0 + logf(s0));
        float logpr = sel[BT + t] - (m1 + logf(s1));
        // coef1 = exp(logp - stop_grad(logp)) == 1 -> -min(adv, adv) = -adv
        float d  = logpr - logp;
        float kl = __expf(d) - d - 1.0f;
        float a  = adv[t / Tt];
        float pt = -a + BETA * kl;
        tokenLoss[t] = pt * (float)amask[t];
    }
}

// -------- final masked-mean reduction --------
__global__ __launch_bounds__(256)
void finalize_kernel(const float* __restrict__ tokenLoss,
                     const int* __restrict__ amask, float* __restrict__ out) {
    __shared__ float sl[256], sm[256];
    float l = 0.0f, mk = 0.0f;
    for (int i = threadIdx.x; i < BT; i += 256) {
        l  += tokenLoss[i];
        mk += (float)amask[i];
    }
    sl[threadIdx.x] = l; sm[threadIdx.x] = mk;
    __syncthreads();
    for (int s = 128; s > 0; s >>= 1) {
        if (threadIdx.x < s) {
            sl[threadIdx.x] += sl[threadIdx.x + s];
            sm[threadIdx.x] += sm[threadIdx.x + s];
        }
        __syncthreads();
    }
    if (threadIdx.x == 0) out[0] = sl[0] / fmaxf(sm[0], 1.0f);
}

extern "C" void kernel_launch(void* const* d_in, const int* in_sizes, int n_in,
                              void* d_out, int out_size, void* d_ws, size_t ws_size,
                              hipStream_t stream) {
    const float* x   = (const float*)d_in[0];   // (B,T,H)
    const float* w   = (const float*)d_in[1];   // (V,H)
    const float* rx  = (const float*)d_in[2];   // (B,T,H)
    const float* rw  = (const float*)d_in[3];   // (V,H)
    const float* adv = (const float*)d_in[4];   // (B,)
    const int*   ids = (const int*)d_in[5];     // (B,T)
    const int*   am  = (const int*)d_in[6];     // (B,T)

    char* ws = (char*)d_ws;
    __bf16* Xb  = (__bf16*)ws; ws += (size_t)BT * Hh * 2;
    __bf16* Rb  = (__bf16*)ws; ws += (size_t)BT * Hh * 2;
    __bf16* Wb  = (__bf16*)ws; ws += (size_t)Vv * Hh * 2;   // 128MB, L2-resident
    __bf16* WRb = (__bf16*)ws; ws += (size_t)Vv * Hh * 2;
    float* pm0 = (float*)ws; ws += (size_t)BT * NSPLIT * 4;
    float* ps0 = (float*)ws; ws += (size_t)BT * NSPLIT * 4;
    float* pm1 = (float*)ws; ws += (size_t)BT * NSPLIT * 4;
    float* ps1 = (float*)ws; ws += (size_t)BT * NSPLIT * 4;
    float* sel = (float*)ws; ws += (size_t)2 * BT * 4;
    float* tl  = (float*)ws; ws += (size_t)BT * 4;

    // 1) precision downconvert: one streaming pass, makes weights L2-resident in bf16
    cvt_f32_to_bf16_kernel<<<2048, 256, 0, stream>>>(x,  Xb,  BT * Hh);
    cvt_f32_to_bf16_kernel<<<2048, 256, 0, stream>>>(rx, Rb,  BT * Hh);
    cvt_f32_to_bf16_kernel<<<8192, 256, 0, stream>>>(w,  Wb,  Vv * Hh);
    cvt_f32_to_bf16_kernel<<<8192, 256, 0, stream>>>(rw, WRb, Vv * Hh);

    // 2) fused LM-head WMMA GEMM + partial softmax (sequential per model for L2 reuse)
    dim3 grid(NBLKN, BT / BLOCK_M);
    lmhead_partial_kernel<<<grid, 256, 0, stream>>>(Xb, Wb,  pm0, ps0);
    lmhead_partial_kernel<<<grid, 256, 0, stream>>>(Rb, WRb, pm1, ps1);

    // 3) selected-token logits (tiny dot products)
    sel_logit_kernel<<<BT / 8, 256, 0, stream>>>(Xb, Wb, Rb, WRb, ids, sel);

    // 4) per-token GRPO loss, 5) masked mean
    combine_kernel<<<BT / 8, 256, 0, stream>>>(pm0, ps0, pm1, ps1, sel, adv, am, tl);
    finalize_kernel<<<1, 256, 0, stream>>>(tl, am, (float*)d_out);
}
